// HierarchicalEncoder_36567351558267
// MI455X (gfx1250) — compile-verified, matrix operand order checked by reading, not run on hardware
//
#include <hip/hip_runtime.h>

// ---------------------------------------------------------------------------
// HierarchicalEncoder for MI455X (gfx1250): f16 WMMA GEMMs + register-resident
// flash attention using transposed-score trick (score D-layout == pT B-layout).
// Round 2: 64x64 GEMM tiles (2x operand reuse), 32-query flash tiles (2x K/V
// reuse), permlanex16 instead of ds_bpermute for cross-half reductions.
// ---------------------------------------------------------------------------

typedef __attribute__((ext_vector_type(16))) _Float16 v16h_t;
typedef __attribute__((ext_vector_type(8)))  _Float16 v8h_t;
typedef __attribute__((ext_vector_type(8)))  float    v8f_t;

#define HE_B  4
#define HE_S  2048
#define HE_D  512
#define HE_H  8
#define HE_F  2048
#define HE_DH 64
#define HE_L  1024   // S/2

// Load an A- or B-fragment for v_wmma_f32_16x16x32_f16.
// Pattern (per 16-bit layout tables): lane = row/col (lane&15), h = lane>>4,
// element e<8  -> K = k0 + 8*h + e
// element e>=8 -> K = k0 + 16 + 8*h + (e-8)
__device__ __forceinline__ v16h_t he_load_frag(const _Float16* __restrict__ row,
                                               int k0, int h) {
  v8h_t lo = *(const v8h_t*)(row + k0 + 8 * h);
  v8h_t hi = *(const v8h_t*)(row + k0 + 16 + 8 * h);
  v16h_t f;
#pragma unroll
  for (int i = 0; i < 8; ++i) { f[i] = lo[i]; f[i + 8] = hi[i]; }
  return f;
}

__device__ __forceinline__ v8f_t he_wmma(v16h_t a, v16h_t b, v8f_t c) {
  return __builtin_amdgcn_wmma_f32_16x16x32_f16(false, a, false, b,
                                                (short)0, c, false, false);
}

__device__ __forceinline__ v8f_t he_zero8() {
  v8f_t z;
#pragma unroll
  for (int i = 0; i < 8; ++i) z[i] = 0.0f;
  return z;
}

// Identity swap between lane halves (== shfl_xor 16 on wave32), VALU-only.
__device__ __forceinline__ float he_xswap(float v) {
  int r = __builtin_amdgcn_permlanex16(__float_as_int(v), __float_as_int(v),
                                       0x76543210, 0xfedcba98, false, false);
  return __int_as_float(r);
}

// ---------------------------------------------------------------------------
// Generic GEMM: out = epilogue(A[M,K](f16) @ W[N,K]^T(f16) + bias)
// one wave per block, 64x64 output tile, 16 WMMAs per K=32 step.
// ---------------------------------------------------------------------------
enum { GEMM_QKV = 0, GEMM_RES32 = 1, GEMM_RELU16 = 2 };

template <int MODE>
__global__ __launch_bounds__(32) void he_gemm(
    const _Float16* __restrict__ A, const _Float16* __restrict__ W,
    const float* __restrict__ bias, int K, int Sq,
    float* __restrict__ out32, const float* __restrict__ res32, int Nout,
    _Float16* __restrict__ o16a, _Float16* __restrict__ o16b,
    _Float16* __restrict__ o16c) {
  const int lane = threadIdx.x;
  const int l = lane & 15, h = lane >> 4;
  const int m0 = blockIdx.x * 64;
  const int n0 = blockIdx.y * 64;

  v8f_t acc[4][4];
#pragma unroll
  for (int i = 0; i < 4; ++i)
#pragma unroll
    for (int j = 0; j < 4; ++j) acc[i][j] = he_zero8();

  const _Float16* arow[4];
#pragma unroll
  for (int i = 0; i < 4; ++i) arow[i] = A + (size_t)(m0 + 16 * i + l) * K;

  for (int k0 = 0; k0 < K; k0 += 32) {
    v16h_t a[4];
#pragma unroll
    for (int i = 0; i < 4; ++i) a[i] = he_load_frag(arow[i], k0, h);
#pragma unroll
    for (int j = 0; j < 4; ++j) {
      v16h_t bf = he_load_frag(W + (size_t)(n0 + 16 * j + l) * K, k0, h);
#pragma unroll
      for (int i = 0; i < 4; ++i) acc[i][j] = he_wmma(a[i], bf, acc[i][j]);
    }
  }

  // C/D layout: lane holds column n = n0+16j+l; rows m = m0+16i + r + 8h.
#pragma unroll
  for (int i = 0; i < 4; ++i) {
#pragma unroll
    for (int j = 0; j < 4; ++j) {
      const int n = n0 + 16 * j + l;
      const float bs = bias[n];
#pragma unroll
      for (int r = 0; r < 8; ++r) {
        const int m = m0 + 16 * i + r + 8 * h;
        float v = acc[i][j][r] + bs;
        if (MODE == GEMM_QKV) {
          const int b = m / Sq, s = m - b * Sq;
          const int which = n >> 9, hh = (n >> 6) & 7, d = n & 63;
          const int bh = b * HE_H + hh;
          if (which == 0)
            o16a[((size_t)bh * Sq + s) * HE_DH + d] = (_Float16)(v * 0.125f);
          else if (which == 1)
            o16b[((size_t)bh * Sq + s) * HE_DH + d] = (_Float16)v;
          else
            o16c[((size_t)bh * HE_DH + d) * Sq + s] = (_Float16)v;   // V^T
        } else if (MODE == GEMM_RES32) {
          out32[(size_t)m * Nout + n] = res32[(size_t)m * Nout + n] + v;
        } else {  // GEMM_RELU16
          o16a[(size_t)m * Nout + n] = (_Float16)fmaxf(v, 0.0f);
        }
      }
    }
  }
}

// ---------------------------------------------------------------------------
// Flash attention, one wave per 32-query tile per (b,h); two 16-row m-tiles
// share every K/V fragment load. Computes s^T = K·Q^T (D-tile: lane=query m,
// regs=key j), online softmax per lane, then o^T = V^T·p^T where p^T's
// B-fragment *is* the score D-registers (f32->f16 convert only).
// ---------------------------------------------------------------------------
__global__ __launch_bounds__(32) void he_flash(
    const _Float16* __restrict__ q, const _Float16* __restrict__ k,
    const _Float16* __restrict__ vT, _Float16* __restrict__ o16, int Sq) {
  const int lane = threadIdx.x, l = lane & 15, h = lane >> 4;
  const int m0 = blockIdx.x * 32;
  const int bh = blockIdx.y;
  const int b = bh / HE_H, hh = bh - b * HE_H;

  const _Float16* qrowA = q + ((size_t)bh * Sq + m0 + l) * HE_DH;
  const _Float16* qrowB = q + ((size_t)bh * Sq + m0 + 16 + l) * HE_DH;
  v16h_t bqA0 = he_load_frag(qrowA, 0, h), bqA1 = he_load_frag(qrowA, 32, h);
  v16h_t bqB0 = he_load_frag(qrowB, 0, h), bqB1 = he_load_frag(qrowB, 32, h);

  v8f_t oA[4], oB[4];
#pragma unroll
  for (int t = 0; t < 4; ++t) { oA[t] = he_zero8(); oB[t] = he_zero8(); }
  float mrunA = -1e30f, lrunA = 0.0f;
  float mrunB = -1e30f, lrunB = 0.0f;

  const _Float16* kbase = k + (size_t)bh * Sq * HE_DH;
  const _Float16* vbase = vT + (size_t)bh * HE_DH * Sq;

  for (int jb = 0; jb < Sq; jb += 32) {
    const _Float16* kr0 = kbase + (size_t)(jb + l) * HE_DH;
    const _Float16* kr1 = kbase + (size_t)(jb + 16 + l) * HE_DH;
    v16h_t a00 = he_load_frag(kr0, 0, h), a01 = he_load_frag(kr0, 32, h);
    v16h_t a10 = he_load_frag(kr1, 0, h), a11 = he_load_frag(kr1, 32, h);

    v8f_t sA0 = he_zero8();
    sA0 = he_wmma(a00, bqA0, sA0); sA0 = he_wmma(a01, bqA1, sA0);
    v8f_t sA1 = he_zero8();
    sA1 = he_wmma(a10, bqA0, sA1); sA1 = he_wmma(a11, bqA1, sA1);
    v8f_t sB0 = he_zero8();
    sB0 = he_wmma(a00, bqB0, sB0); sB0 = he_wmma(a01, bqB1, sB0);
    v8f_t sB1 = he_zero8();
    sB1 = he_wmma(a10, bqB0, sB1); sB1 = he_wmma(a11, bqB1, sB1);

    // ---- m-tile A softmax update ----
    float mt = -1e30f;
#pragma unroll
    for (int r = 0; r < 8; ++r) { mt = fmaxf(mt, sA0[r]); mt = fmaxf(mt, sA1[r]); }
    mt = fmaxf(mt, he_xswap(mt));
    float mnew = fmaxf(mrunA, mt);
    float alpha = __expf(mrunA - mnew);
    float ps = 0.0f;
    v16h_t pbA;
#pragma unroll
    for (int r = 0; r < 8; ++r) {
      float p0 = __expf(sA0[r] - mnew);
      float p1 = __expf(sA1[r] - mnew);
      ps += p0 + p1;
      pbA[r] = (_Float16)p0;
      pbA[r + 8] = (_Float16)p1;
    }
    lrunA = lrunA * alpha + ps;
    mrunA = mnew;
#pragma unroll
    for (int t = 0; t < 4; ++t)
#pragma unroll
      for (int r = 0; r < 8; ++r) oA[t][r] *= alpha;

    // ---- m-tile B softmax update ----
    mt = -1e30f;
#pragma unroll
    for (int r = 0; r < 8; ++r) { mt = fmaxf(mt, sB0[r]); mt = fmaxf(mt, sB1[r]); }
    mt = fmaxf(mt, he_xswap(mt));
    mnew = fmaxf(mrunB, mt);
    alpha = __expf(mrunB - mnew);
    ps = 0.0f;
    v16h_t pbB;
#pragma unroll
    for (int r = 0; r < 8; ++r) {
      float p0 = __expf(sB0[r] - mnew);
      float p1 = __expf(sB1[r] - mnew);
      ps += p0 + p1;
      pbB[r] = (_Float16)p0;
      pbB[r + 8] = (_Float16)p1;
    }
    lrunB = lrunB * alpha + ps;
    mrunB = mnew;
#pragma unroll
    for (int t = 0; t < 4; ++t)
#pragma unroll
      for (int r = 0; r < 8; ++r) oB[t][r] *= alpha;

    // ---- o^T += V^T tile x p^T (V fragments shared by both m-tiles) ----
#pragma unroll
    for (int t = 0; t < 4; ++t) {
      v16h_t av = he_load_frag(vbase + (size_t)(t * 16 + l) * Sq, jb, h);
      oA[t] = he_wmma(av, pbA, oA[t]);
      oB[t] = he_wmma(av, pbB, oB[t]);
    }
  }

  const float invA = 1.0f / (lrunA + he_xswap(lrunA));
  const float invB = 1.0f / (lrunB + he_xswap(lrunB));
  _Float16* orowA = o16 + ((size_t)b * Sq + m0 + l) * HE_D + hh * HE_DH;
  _Float16* orowB = o16 + ((size_t)b * Sq + m0 + 16 + l) * HE_D + hh * HE_DH;
#pragma unroll
  for (int t = 0; t < 4; ++t)
#pragma unroll
    for (int r = 0; r < 8; ++r) {
      orowA[t * 16 + r + 8 * h] = (_Float16)(oA[t][r] * invA);
      orowB[t * 16 + r + 8 * h] = (_Float16)(oB[t][r] * invB);
    }
}

// ---------------------------------------------------------------------------
// LayerNorm over D=512, one row per 256-thread block; writes f32 + f16.
// ---------------------------------------------------------------------------
__global__ __launch_bounds__(256) void he_ln(const float* __restrict__ xin,
                                             const float* __restrict__ g,
                                             const float* __restrict__ be,
                                             float* __restrict__ o32,
                                             _Float16* __restrict__ o16) {
  const int t = threadIdx.x;
  const float* xr = xin + (size_t)blockIdx.x * HE_D;
  __shared__ float sh[256];
  const float a = xr[t], c = xr[t + 256];
  sh[t] = a + c;
  __syncthreads();
  for (int off = 128; off > 0; off >>= 1) {
    if (t < off) sh[t] += sh[t + off];
    __syncthreads();
  }
  const float mu = sh[0] * (1.0f / HE_D);
  __syncthreads();
  const float d0 = a - mu, d1 = c - mu;
  sh[t] = d0 * d0 + d1 * d1;
  __syncthreads();
  for (int off = 128; off > 0; off >>= 1) {
    if (t < off) sh[t] += sh[t + off];
    __syncthreads();
  }
  const float rstd = rsqrtf(sh[0] * (1.0f / HE_D) + 1e-5f);
  const float y0 = d0 * rstd * g[t] + be[t];
  const float y1 = d1 * rstd * g[t + 256] + be[t + 256];
  float* orow = o32 + (size_t)blockIdx.x * HE_D;
  _Float16* hrow = o16 + (size_t)blockIdx.x * HE_D;
  orow[t] = y0;        orow[t + 256] = y1;
  hrow[t] = (_Float16)y0; hrow[t + 256] = (_Float16)y1;
}

// ---------------------------------------------------------------------------
// Segment starts from boundary probs (serial per batch; segments are runs).
// seg[j] = #{1<=i<j : bp[i]>0.5}; boundary at i => new segment starts at i+1.
// Only i in [1, S-2] affect seg; n_chunks = count + 1.
// ---------------------------------------------------------------------------
__global__ void he_seg(const float* __restrict__ bp, int* __restrict__ starts,
                       int* __restrict__ nch) {
  const int b = threadIdx.x;
  if (b >= HE_B) return;
  const float* p = bp + (size_t)b * HE_S;
  int* st = starts + (size_t)b * (HE_S + 1);
  int cur = 0;
  st[0] = 0;
  for (int i = 1; i <= HE_S - 2; ++i) {
    if (p[i] > 0.5f) { ++cur; st[cur] = i + 1; }
  }
  const int n = cur + 1;
  st[n] = HE_S;
  nch[b] = n;
}

__global__ __launch_bounds__(64) void he_chunk_mean(
    const float* __restrict__ x32, const int* __restrict__ starts,
    const int* __restrict__ nch, float* __restrict__ c32,
    _Float16* __restrict__ c16) {
  const int c = blockIdx.x, b = blockIdx.y, t = threadIdx.x;
  float acc[8];
#pragma unroll
  for (int qn = 0; qn < 8; ++qn) acc[qn] = 0.0f;
  const int n = nch[b];
  int j0 = 0, j1 = 0;
  float scale = 0.0f;
  if (c < n) {
    j0 = starts[(size_t)b * (HE_S + 1) + c];
    j1 = starts[(size_t)b * (HE_S + 1) + c + 1];
    scale = 1.0f / (float)(j1 - j0);
  }
  const float* xb = x32 + (size_t)b * HE_S * HE_D;
  for (int j = j0; j < j1; ++j) {
    const float* xr = xb + (size_t)j * HE_D;
#pragma unroll
    for (int qn = 0; qn < 8; ++qn) acc[qn] += xr[t + 64 * qn];
  }
  const size_t orow = ((size_t)b * HE_L + c) * HE_D;
#pragma unroll
  for (int qn = 0; qn < 8; ++qn) {
    const float mv = acc[qn] * scale;
    c32[orow + t + 64 * qn] = mv;
    c16[orow + t + 64 * qn] = (_Float16)mv;
  }
}

__global__ void he_f32_to_f16(const float* __restrict__ a,
                              _Float16* __restrict__ o, long long n) {
  long long i = (long long)blockIdx.x * blockDim.x + threadIdx.x;
  if (i < n) o[i] = (_Float16)a[i];
}

// ---------------------------------------------------------------------------
extern "C" void kernel_launch(void* const* d_in, const int* in_sizes, int n_in,
                              void* d_out, int out_size, void* d_ws,
                              size_t ws_size, hipStream_t stream) {
  (void)in_sizes; (void)n_in; (void)out_size; (void)ws_size;
  const float* x    = (const float*)d_in[0];
  const float* bp   = (const float*)d_in[1];
  const float* Wqkv = (const float*)d_in[2];
  const float* bqkv = (const float*)d_in[3];
  const float* Wo   = (const float*)d_in[4];
  const float* bo   = (const float*)d_in[5];
  const float* W1   = (const float*)d_in[6];
  const float* b1   = (const float*)d_in[7];
  const float* W2   = (const float*)d_in[8];
  const float* b2   = (const float*)d_in[9];
  const float* g1   = (const float*)d_in[10];
  const float* be1  = (const float*)d_in[11];
  const float* g2   = (const float*)d_in[12];
  const float* be2  = (const float*)d_in[13];
  float* out = (float*)d_out;

  const size_t Mb = (size_t)HE_B * HE_S;   // 8192 byte-stage tokens
  const size_t Mc = (size_t)HE_B * HE_L;   // 4096 chunk-stage tokens

  char* p = (char*)d_ws;
  auto bump = [&](size_t bytes) -> char* {
    char* r = p;
    p += (bytes + 255) & ~(size_t)255;
    return r;
  };
  _Float16* wqkv16 = (_Float16*)bump((size_t)4 * 1536 * 512 * 2);
  _Float16* wo16   = (_Float16*)bump((size_t)4 * 512 * 512 * 2);
  _Float16* w116   = (_Float16*)bump((size_t)4 * 2048 * 512 * 2);
  _Float16* w216   = (_Float16*)bump((size_t)4 * 512 * 2048 * 2);
  _Float16* x16 = (_Float16*)bump(Mb * HE_D * 2);
  float*    x32 = (float*)bump(Mb * HE_D * 4);
  _Float16* q16 = (_Float16*)bump(Mb * HE_D * 2);
  _Float16* k16 = (_Float16*)bump(Mb * HE_D * 2);
  _Float16* vT16 = (_Float16*)bump(Mb * HE_D * 2);
  _Float16* o16 = (_Float16*)bump(Mb * HE_D * 2);
  _Float16* h16 = (_Float16*)bump(Mb * HE_F * 2);
  float*    t32 = (float*)bump(Mb * HE_D * 4);
  float*    y32 = (float*)bump(Mb * HE_D * 4);
  _Float16* y16 = (_Float16*)bump(Mb * HE_D * 2);
  float*    c32 = (float*)bump(Mc * HE_D * 4);
  _Float16* c16 = (_Float16*)bump(Mc * HE_D * 2);
  int* starts = (int*)bump((size_t)HE_B * (HE_S + 1) * 4);
  int* nch    = (int*)bump((size_t)HE_B * 4);

  auto conv = [&](const float* s, _Float16* d, size_t n) {
    he_f32_to_f16<<<dim3((unsigned)((n + 255) / 256)), 256, 0, stream>>>(
        s, d, (long long)n);
  };
  conv(Wqkv, wqkv16, (size_t)4 * 1536 * 512);
  conv(Wo, wo16, (size_t)4 * 512 * 512);
  conv(W1, w116, (size_t)4 * 2048 * 512);
  conv(W2, w216, (size_t)4 * 512 * 2048);
  conv(x, x16, Mb * HE_D);

  // ---------------- byte encoder: layers 0,1 on [B*S, D] ----------------
  for (int lyr = 0; lyr < 2; ++lyr) {
    const float* resid = (lyr == 0) ? x : x32;
    he_gemm<GEMM_QKV><<<dim3((unsigned)(Mb / 64), 1536 / 64), 32, 0, stream>>>(
        x16, wqkv16 + (size_t)lyr * 1536 * 512, bqkv + lyr * 1536, 512, HE_S,
        nullptr, nullptr, 0, q16, k16, vT16);
    he_flash<<<dim3(HE_S / 32, HE_B * HE_H), 32, 0, stream>>>(q16, k16, vT16,
                                                              o16, HE_S);
    he_gemm<GEMM_RES32><<<dim3((unsigned)(Mb / 64), 512 / 64), 32, 0, stream>>>(
        o16, wo16 + (size_t)lyr * 512 * 512, bo + lyr * 512, 512, HE_S,
        t32, resid, 512, nullptr, nullptr, nullptr);
    he_ln<<<dim3((unsigned)Mb), 256, 0, stream>>>(t32, g1 + lyr * 512,
                                                  be1 + lyr * 512, y32, y16);
    he_gemm<GEMM_RELU16><<<dim3((unsigned)(Mb / 64), 2048 / 64), 32, 0, stream>>>(
        y16, w116 + (size_t)lyr * 2048 * 512, b1 + lyr * 2048, 512, HE_S,
        nullptr, nullptr, 2048, h16, nullptr, nullptr);
    he_gemm<GEMM_RES32><<<dim3((unsigned)(Mb / 64), 512 / 64), 32, 0, stream>>>(
        h16, w216 + (size_t)lyr * 512 * 2048, b2 + lyr * 512, 2048, HE_S,
        t32, y32, 512, nullptr, nullptr, nullptr);
    he_ln<<<dim3((unsigned)Mb), 256, 0, stream>>>(t32, g2 + lyr * 512,
                                                  be2 + lyr * 512, x32, x16);
  }

  // ---------------- chunk pooling ----------------
  he_seg<<<1, 32, 0, stream>>>(bp, starts, nch);
  he_chunk_mean<<<dim3(HE_L, HE_B), 64, 0, stream>>>(x32, starts, nch, c32, c16);

  // ---------------- chunk encoder: layers 2,3 on [B*L, D] ----------------
  for (int lyr = 2; lyr < 4; ++lyr) {
    const float* resid = c32;
    he_gemm<GEMM_QKV><<<dim3((unsigned)(Mc / 64), 1536 / 64), 32, 0, stream>>>(
        c16, wqkv16 + (size_t)lyr * 1536 * 512, bqkv + lyr * 1536, 512, HE_L,
        nullptr, nullptr, 0, q16, k16, vT16);
    he_flash<<<dim3(HE_L / 32, HE_B * HE_H), 32, 0, stream>>>(q16, k16, vT16,
                                                              o16, HE_L);
    he_gemm<GEMM_RES32><<<dim3((unsigned)(Mc / 64), 512 / 64), 32, 0, stream>>>(
        o16, wo16 + (size_t)lyr * 512 * 512, bo + lyr * 512, 512, HE_L,
        t32, resid, 512, nullptr, nullptr, nullptr);
    he_ln<<<dim3((unsigned)Mc), 256, 0, stream>>>(t32, g1 + lyr * 512,
                                                  be1 + lyr * 512, y32, y16);
    he_gemm<GEMM_RELU16><<<dim3((unsigned)(Mc / 64), 2048 / 64), 32, 0, stream>>>(
        y16, w116 + (size_t)lyr * 2048 * 512, b1 + lyr * 2048, 512, HE_L,
        nullptr, nullptr, 2048, h16, nullptr, nullptr);
    he_gemm<GEMM_RES32><<<dim3((unsigned)(Mc / 64), 512 / 64), 32, 0, stream>>>(
        h16, w216 + (size_t)lyr * 512 * 2048, b2 + lyr * 512, 2048, HE_L,
        t32, y32, 512, nullptr, nullptr, nullptr);
    float* ox = (lyr == 3) ? out : c32;
    he_ln<<<dim3((unsigned)Mc), 256, 0, stream>>>(t32, g2 + lyr * 512,
                                                  be2 + lyr * 512, ox, c16);
  }
}
